// SymbolAttention_3607772529021
// MI455X (gfx1250) — matrix operand order repaired and checked by reading, not compiled, not run on hardware
//
#include <hip/hip_runtime.h>

// ---------------------------------------------------------------------------
// Problem constants (from the reference): B=4, S=2048, D=1024, H=16, DH=64
// ---------------------------------------------------------------------------
#define BATCH 4
#define SEQ   2048
#define DMOD  1024
#define NHEAD 16
#define DHEAD 64
#define BS    (BATCH * SEQ)   // 8192 total rows

typedef __attribute__((ext_vector_type(16))) __bf16 v16bf;
typedef __attribute__((ext_vector_type(8)))  float  v8f;

union FragU {
    uint4  u[2];
    v16bf  v;
};

__device__ __forceinline__ v16bf mkfrag(uint4 lo, uint4 hi) {
    FragU f;
    f.u[0] = lo;
    f.u[1] = hi;
    return f.v;
}

__device__ __forceinline__ v8f wmma_bf16(v16bf a, v16bf b, v8f c) {
    // D = A(16x32 bf16) * B(32x16 bf16) + C(16x16 f32)
    return __builtin_amdgcn_wmma_f32_16x16x32_bf16(
        false, a, false, b, (short)0, c, false, false);
}

// ---------------------------------------------------------------------------
// f32 -> bf16 elementwise convert, 4 elements/thread (pure-BW kernel)
// ---------------------------------------------------------------------------
__global__ __launch_bounds__(256) void cvt_f32_bf16_v4(
    const float4* __restrict__ src, uint2* __restrict__ dst, int n4) {
    int i = blockIdx.x * 256 + threadIdx.x;
    if (i < n4) {
        float4 f = src[i];
        union { __bf16 h[4]; uint2 u; } cv;
        cv.h[0] = (__bf16)f.x;
        cv.h[1] = (__bf16)f.y;
        cv.h[2] = (__bf16)f.z;
        cv.h[3] = (__bf16)f.w;
        dst[i] = cv.u;
    }
}

// ---------------------------------------------------------------------------
// Batched LDS-tiled transpose + convert to bf16:
//   dst[z][c*R + r] = (bf16) src[z][r*C + c]
// grid = (C/32, R/32, nbatch), block = (32, 8)
// ---------------------------------------------------------------------------
template <typename T>
__global__ __launch_bounds__(256) void transpose_to_bf16(
    const T* __restrict__ src, __bf16* __restrict__ dst, int R, int C) {
    __shared__ float tile[32][33];
    const size_t bofs = (size_t)blockIdx.z * (size_t)R * (size_t)C;
    src += bofs;
    dst += bofs;
    const int c0 = blockIdx.x * 32;
    const int r0 = blockIdx.y * 32;
    for (int i = threadIdx.y; i < 32; i += 8)
        tile[i][threadIdx.x] = (float)src[(size_t)(r0 + i) * C + c0 + threadIdx.x];
    __syncthreads();
    for (int i = threadIdx.y; i < 32; i += 8)
        dst[(size_t)(c0 + i) * R + r0 + threadIdx.x] = (__bf16)tile[threadIdx.x][i];
}

// ---------------------------------------------------------------------------
// Generic bf16 WMMA GEMM:  C[z] = A[z] (MxK, row major, lda)
//                                * Bt[z]^T (Bt is NxK row major, ldb)
// Output f32 or bf16 (out_bf16 flag). M, N multiples of 128; K mult of 32.
// grid = (N/128, M/128, nbatch), block = 128 (4 waves, 2x2 wave grid).
// Each wave computes a 64x64 tile as 4x4 WMMA subtiles: 16 b128 loads feed
// 16 WMMAs per K-step (1 load per WMMA, vs 2 with 2x2 tiling).
// ---------------------------------------------------------------------------
__global__ __launch_bounds__(128) void gemm_bf16_wmma(
    const __bf16* __restrict__ A, int lda, long long sA,
    const __bf16* __restrict__ Bt, int ldb, long long sB,
    void* __restrict__ C, int ldc, long long sC,
    int Kdim, int out_bf16) {
    const int z = blockIdx.z;
    A  += (size_t)z * (size_t)sA;
    Bt += (size_t)z * (size_t)sB;

    const int wave = threadIdx.x >> 5;
    const int lane = threadIdx.x & 31;
    const int l    = lane & 15;
    const int half = lane >> 4;

    const int m0 = blockIdx.y * 128 + (wave >> 1) * 64;
    const int n0 = blockIdx.x * 128 + (wave & 1) * 64;

    v8f acc[4][4];
    for (int mi = 0; mi < 4; ++mi)
        for (int ni = 0; ni < 4; ++ni)
            acc[mi][ni] = {};

    for (int k0 = 0; k0 < Kdim; k0 += 32) {
        // A fragments (16x32): lane row = m, halves split K per ISA layout
        v16bf af[4];
        for (int mi = 0; mi < 4; ++mi) {
            const __bf16* p = A + (size_t)(m0 + mi * 16 + l) * lda + k0 + half * 8;
            af[mi] = mkfrag(*(const uint4*)p, *(const uint4*)(p + 16));
            __builtin_prefetch(p + 64, 0, 1);  // next-K cacheline -> global_prefetch
        }
        // B fragments (32x16): lane = column n of B = row of Bt, K contiguous
        for (int ni = 0; ni < 4; ++ni) {
            const __bf16* p = Bt + (size_t)(n0 + ni * 16 + l) * ldb + k0 + half * 8;
            v16bf bfr = mkfrag(*(const uint4*)p, *(const uint4*)(p + 16));
            __builtin_prefetch(p + 64, 0, 1);
            for (int mi = 0; mi < 4; ++mi)
                acc[mi][ni] = wmma_bf16(af[mi], bfr, acc[mi][ni]);
        }
    }

    // D layout: VGPR j, lanes0-15 -> m=j, lanes16-31 -> m=j+8; n = lane%16
    if (out_bf16) {
        __bf16* Cp = (__bf16*)C + (size_t)z * (size_t)sC;
        for (int mi = 0; mi < 4; ++mi)
            for (int ni = 0; ni < 4; ++ni)
                for (int j = 0; j < 8; ++j) {
                    int m = m0 + mi * 16 + j + half * 8;
                    int n = n0 + ni * 16 + l;
                    Cp[(size_t)m * ldc + n] = (__bf16)acc[mi][ni][j];
                }
    } else {
        float* Cp = (float*)C + (size_t)z * (size_t)sC;
        for (int mi = 0; mi < 4; ++mi)
            for (int ni = 0; ni < 4; ++ni)
                for (int j = 0; j < 8; ++j) {
                    int m = m0 + mi * 16 + j + half * 8;
                    int n = n0 + ni * 16 + l;
                    Cp[(size_t)m * ldc + n] = acc[mi][ni][j];
                }
    }
}

// ---------------------------------------------------------------------------
// Flash attention (per batch-head), bf16 WMMA, online softmax.
//   QM : [B, S, D] bf16 (head h at column h*DHEAD)
//   Kb : [B, S, D] bf16
//   Vt : [B*H, DHEAD, S] bf16 (V transposed per head)
//   mask: [S, S] f32 additive
//   O  : [B, S, D] bf16 (merged heads)
// grid = (S/64, B*H), block = 128: each wave owns 16 query rows.
// scores = clip(QK^T / sqrt(DH), +-50) + mask; softmax; out = P @ V
// ---------------------------------------------------------------------------
__global__ __launch_bounds__(128) void flash_attn_wmma(
    const __bf16* __restrict__ QM, const __bf16* __restrict__ Kb,
    const __bf16* __restrict__ Vt, const float* __restrict__ mask,
    __bf16* __restrict__ O) {
    const int bh   = blockIdx.y;
    const int b    = bh / NHEAD;
    const int h    = bh % NHEAD;
    const int wave = threadIdx.x >> 5;
    const int lane = threadIdx.x & 31;
    const int l    = lane & 15;
    const int half = lane >> 4;
    const int qbase = blockIdx.x * 64 + wave * 16;

    // Per-wave P staging buffer: 16 rows x 64 cols bf16, row stride 80 (=160B,
    // multiple of 16B so b128 LDS accesses stay aligned).
    __shared__ __align__(16) __bf16 pbuf[4][16][80];

    // Hoisted Q fragments: 16 rows x 64 (two K-steps of 32)
    const __bf16* qrow = QM + ((size_t)b * SEQ + qbase + l) * DMOD + h * DHEAD;
    v16bf qfrag[2];
    for (int ks = 0; ks < 2; ++ks) {
        const __bf16* p = qrow + ks * 32 + half * 8;
        qfrag[ks] = mkfrag(*(const uint4*)p, *(const uint4*)(p + 16));
    }

    const __bf16* kbase = Kb + (size_t)b * SEQ * DMOD + h * DHEAD;
    const __bf16* vbase = Vt + (size_t)bh * DHEAD * SEQ;

    float m_run[8], l_run[8];
    v8f   oacc[4];
    for (int j = 0; j < 8; ++j) { m_run[j] = -1e30f; l_run[j] = 0.0f; }
    for (int di = 0; di < 4; ++di) oacc[di] = {};

    for (int j0 = 0; j0 < SEQ; j0 += 64) {
        // ---- S tile = Q * K^T  (16 x 64) ----
        v8f sacc[4];
        for (int ni = 0; ni < 4; ++ni) sacc[ni] = {};
        for (int ks = 0; ks < 2; ++ks) {
            for (int ni = 0; ni < 4; ++ni) {
                const __bf16* kp =
                    kbase + (size_t)(j0 + ni * 16 + l) * DMOD + ks * 32 + half * 8;
                v16bf kf = mkfrag(*(const uint4*)kp, *(const uint4*)(kp + 16));
                if (ks == 0)  // prefetch next KV tile's K rows
                    __builtin_prefetch(kp + (size_t)64 * DMOD, 0, 1);
                sacc[ni] = wmma_bf16(qfrag[ks], kf, sacc[ni]);
            }
        }

        // ---- scale (1/sqrt(64)=0.125), clip +-50, additive mask ----
        for (int ni = 0; ni < 4; ++ni)
            for (int j = 0; j < 8; ++j) {
                float s = sacc[ni][j] * 0.125f;
                s = fminf(fmaxf(s, -50.0f), 50.0f);
                const int q = qbase + j + half * 8;
                const int t = j0 + ni * 16 + l;
                s += mask[(size_t)q * SEQ + t];
                sacc[ni][j] = s;
            }

        // ---- online softmax (rows live in 16-lane halves on wave32) ----
        float corr[8];
        for (int j = 0; j < 8; ++j) {
            float mx = fmaxf(fmaxf(sacc[0][j], sacc[1][j]),
                             fmaxf(sacc[2][j], sacc[3][j]));
            for (int off = 8; off >= 1; off >>= 1)
                mx = fmaxf(mx, __shfl_xor(mx, off, 16));
            const float mnew = fmaxf(m_run[j], mx);
            corr[j] = __expf(m_run[j] - mnew);
            float rsum = 0.0f;
            for (int ni = 0; ni < 4; ++ni) {
                float p = __expf(sacc[ni][j] - mnew);
                sacc[ni][j] = p;
                rsum += p;
            }
            for (int off = 8; off >= 1; off >>= 1)
                rsum += __shfl_xor(rsum, off, 16);
            l_run[j] = l_run[j] * corr[j] + rsum;
            m_run[j] = mnew;
            for (int di = 0; di < 4; ++di) oacc[di][j] *= corr[j];
        }

        // ---- re-layout P via LDS: D-fragment -> A-fragment ----
        __syncthreads();  // previous iteration's readers are done
        for (int ni = 0; ni < 4; ++ni)
            for (int j = 0; j < 8; ++j)
                pbuf[wave][j + half * 8][ni * 16 + l] = (__bf16)sacc[ni][j];
        __syncthreads();  // writes visible before reads

        // ---- O += P * V  (K = 64 in two 32-steps) ----
        for (int ks = 0; ks < 2; ++ks) {
            const __bf16* pr = &pbuf[wave][l][ks * 32 + half * 8];
            v16bf pf = mkfrag(*(const uint4*)pr, *(const uint4*)(pr + 16));
            for (int di = 0; di < 4; ++di) {
                const __bf16* vp = vbase + (size_t)(di * 16 + l) * SEQ + j0 +
                                   ks * 32 + half * 8;
                v16bf vf = mkfrag(*(const uint4*)vp, *(const uint4*)(vp + 16));
                oacc[di] = wmma_bf16(pf, vf, oacc[di]);
            }
        }
    }

    // ---- epilogue: O /= l, write merged-head bf16 output ----
    float invl[8];
    for (int j = 0; j < 8; ++j) invl[j] = 1.0f / l_run[j];
    for (int di = 0; di < 4; ++di)
        for (int j = 0; j < 8; ++j) {
            const int q = qbase + j + half * 8;
            const int d = di * 16 + l;
            O[((size_t)b * SEQ + q) * DMOD + h * DHEAD + d] =
                (__bf16)(oacc[di][j] * invl[j]);
        }
}

// ---------------------------------------------------------------------------
// Host-side orchestration
// Inputs (setup_inputs order): x, M, mask, Wq, Wk, Wv, Wo  (all f32)
// Output: [B, S, D] f32
// Workspace layout (128 MB total):
//   0   xb [BS,D]        16 MB      | 48  kb  [BS,D]      16 MB
//   16  wqb [D,D]         2 MB      | 64  vb  [BS,D]      16 MB
//   18  wkb               2 MB      | 80  vtb [B*H,DH,S]  16 MB
//   20  wvb               2 MB      | 96  qmb [BS,D]      16 MB
//   22  wob               2 MB      | 112 ob  [BS,D]      16 MB
//   24  mtb [B,D,D]       8 MB      |
//   32  qb  [BS,D]       16 MB      |
// ---------------------------------------------------------------------------
extern "C" void kernel_launch(void* const* d_in, const int* in_sizes, int n_in,
                              void* d_out, int out_size, void* d_ws,
                              size_t ws_size, hipStream_t stream) {
    const float* x    = (const float*)d_in[0];
    const float* Mm   = (const float*)d_in[1];
    const float* mask = (const float*)d_in[2];
    const float* Wq   = (const float*)d_in[3];
    const float* Wk   = (const float*)d_in[4];
    const float* Wv   = (const float*)d_in[5];
    const float* Wo   = (const float*)d_in[6];
    float*       out  = (float*)d_out;

    char* w = (char*)d_ws;
    const size_t MB = 1024 * 1024;
    __bf16* xb  = (__bf16*)(w + 0 * MB);
    __bf16* wqb = (__bf16*)(w + 16 * MB);
    __bf16* wkb = (__bf16*)(w + 18 * MB);
    __bf16* wvb = (__bf16*)(w + 20 * MB);
    __bf16* wob = (__bf16*)(w + 22 * MB);
    __bf16* mtb = (__bf16*)(w + 24 * MB);
    __bf16* qb  = (__bf16*)(w + 32 * MB);
    __bf16* kb  = (__bf16*)(w + 48 * MB);
    __bf16* vb  = (__bf16*)(w + 64 * MB);
    __bf16* vtb = (__bf16*)(w + 80 * MB);
    __bf16* qmb = (__bf16*)(w + 96 * MB);
    __bf16* ob  = (__bf16*)(w + 112 * MB);

    const int nx = BS * DMOD;       // 8,388,608
    const int nw = DMOD * DMOD;     // 1,048,576

    // 1) converts (vectorized x4)
    cvt_f32_bf16_v4<<<(nx / 4 + 255) / 256, 256, 0, stream>>>(
        (const float4*)x, (uint2*)xb, nx / 4);
    cvt_f32_bf16_v4<<<(nw / 4 + 255) / 256, 256, 0, stream>>>(
        (const float4*)Wq, (uint2*)wqb, nw / 4);
    cvt_f32_bf16_v4<<<(nw / 4 + 255) / 256, 256, 0, stream>>>(
        (const float4*)Wk, (uint2*)wkb, nw / 4);
    cvt_f32_bf16_v4<<<(nw / 4 + 255) / 256, 256, 0, stream>>>(
        (const float4*)Wv, (uint2*)wvb, nw / 4);
    cvt_f32_bf16_v4<<<(nw / 4 + 255) / 256, 256, 0, stream>>>(
        (const float4*)Wo, (uint2*)wob, nw / 4);

    // 2) M^T per batch (QM = Q @ M needs B-operand as [e, d])
    transpose_to_bf16<float><<<dim3(DMOD / 32, DMOD / 32, BATCH),
                               dim3(32, 8), 0, stream>>>(Mm, mtb, DMOD, DMOD);

    // 3) Q/K/V projections: y = x @ W^T  (row-major W is exactly Bt)
    dim3 gproj(DMOD / 128, BS / 128, 1);
    gemm_bf16_wmma<<<gproj, 128, 0, stream>>>(xb, DMOD, 0, wqb, DMOD, 0,
                                              qb, DMOD, 0, DMOD, 1);
    gemm_bf16_wmma<<<gproj, 128, 0, stream>>>(xb, DMOD, 0, wkb, DMOD, 0,
                                              kb, DMOD, 0, DMOD, 1);
    gemm_bf16_wmma<<<gproj, 128, 0, stream>>>(xb, DMOD, 0, wvb, DMOD, 0,
                                              vb, DMOD, 0, DMOD, 1);

    // 4) V^T per batch: [S, D] -> [D, S]  (== per-head [DH, S] contiguous)
    transpose_to_bf16<__bf16><<<dim3(DMOD / 32, SEQ / 32, BATCH),
                                dim3(32, 8), 0, stream>>>(vb, vtb, SEQ, DMOD);

    // 5) QM = Q @ M (batched over B, Bt = M^T)
    gemm_bf16_wmma<<<dim3(DMOD / 128, SEQ / 128, BATCH), 128, 0, stream>>>(
        qb, DMOD, (long long)SEQ * DMOD, mtb, DMOD, (long long)DMOD * DMOD,
        qmb, DMOD, (long long)SEQ * DMOD, DMOD, 1);

    // 6) flash attention over (b, h)
    flash_attn_wmma<<<dim3(SEQ / 64, BATCH * NHEAD), 128, 0, stream>>>(
        qmb, kb, vtb, mask, ob);

    // 7) final projection: out = O @ Wo^T  (f32 output)
    gemm_bf16_wmma<<<gproj, 128, 0, stream>>>(ob, DMOD, 0, wob, DMOD, 0,
                                              out, DMOD, 0, DMOD, 0);
}
